// LlamaAttentionMorph_77988016161103
// MI455X (gfx1250) — compile-verified, matrix operand order checked by reading, not run on hardware
//
#include <hip/hip_runtime.h>
#include <hip/hip_bf16.h>
#include <cstdint>

// ---------------------------------------------------------------------------
// Problem constants (match reference): B=1, S=2048, DM=4096, H=32, KV=8, D=128
// ---------------------------------------------------------------------------
#define S_LEN   2048
#define DMODEL  4096
#define NH      32
#define NKV     8
#define HD      128
#define NQKV    6144   // NH*HD + 2*NKV*HD

typedef __attribute__((ext_vector_type(16))) __bf16       bf16x16;
typedef __attribute__((ext_vector_type(8)))  float        f32x8;
typedef __attribute__((ext_vector_type(4)))  float        f32x4;
typedef __attribute__((ext_vector_type(4)))  unsigned int u32x4;

__device__ __forceinline__ f32x8 zero8() {
    f32x8 z;
    for (int i = 0; i < 8; ++i) z[i] = 0.f;
    return z;
}

// D = A(16x32 bf16) x B(32x16 bf16) + C(16x16 f32), wave32 WMMA.
__device__ __forceinline__ f32x8 wmma_bf16(bf16x16 a, bf16x16 b, f32x8 c) {
    return __builtin_amdgcn_wmma_f32_16x16x32_bf16(
        /*neg_a=*/false, a, /*neg_b=*/false, b,
        /*c_mod=*/(short)0, c, /*reuse_a=*/false, /*reuse_b=*/false);
}

// Load one 16-element bf16 fragment for a lane. ISA 16-bit A/B layout:
// elem e <-> k = (e/8)*16 + half*8 + e%8, i.e. two contiguous 16-byte runs.
__device__ __forceinline__ bf16x16 ld_frag(const __bf16* row32, int half) {
    union { bf16x16 v; u32x4 u[2]; } t;
    t.u[0] = *(const u32x4*)(row32 + half * 8);
    t.u[1] = *(const u32x4*)(row32 + 16 + half * 8);
    return t.v;
}

// CDNA5 async global->LDS copy (GVS mode: saddr base + 32-bit vgpr offset).
// Tracked with ASYNCcnt; completes in order with other async loads.
__device__ __forceinline__ void async_ld_b128(uint32_t lds_byte_off,
                                              const void* sbase,
                                              uint32_t voff_bytes) {
    asm volatile("global_load_async_to_lds_b128 %0, %1, %2 offset:0"
                 :: "v"(lds_byte_off), "v"(voff_bytes), "s"(sbase)
                 : "memory");
}

// ---------------------------------------------------------------------------
// GEMM: C[M x N](f32) = A[M x K] * B[K x N](f32 row-major, ldb)
// A is f32 (converted to bf16 while staging) or already bf16 (template).
// Block tile 128x128, BK=32, 256 threads = 8 waves (2x4), wave tile 64x32.
// Double-buffered LDS: prefetch tile k+1 globals while tile k runs WMMA.
// ---------------------------------------------------------------------------
template <bool A_BF16>
__global__ __launch_bounds__(256) void gemm_kernel(
    const void* __restrict__ Av, const float* __restrict__ Bp,
    float* __restrict__ Cp, int K, int ldb, int ldc) {
    __shared__ __bf16 As[2][128][32];   // [buf][m][k]
    __shared__ __bf16 Bs[2][128][32];   // [buf][n][k] (transposed staging)

    const int tid  = threadIdx.x;
    const int lane = tid & 31;
    const int wid  = tid >> 5;
    const int half = lane >> 4;
    const int ln16 = lane & 15;
    const int wm   = wid >> 2;          // 0..1
    const int wn   = wid & 3;           // 0..3
    const int m0   = blockIdx.y * 128;
    const int n0   = blockIdx.x * 128;

    auto stage = [&](int k0, int buf) {
        if (A_BF16) {
            const __bf16* A = (const __bf16*)Av;
            for (int s = tid; s < 512; s += 256) {        // 8 bf16 per slot
                int r = s >> 2, c = (s & 3) << 3;
                *(u32x4*)&As[buf][r][c] =
                    *(const u32x4*)(A + (size_t)(m0 + r) * K + k0 + c);
            }
        } else {
            const float* A = (const float*)Av;
            for (int s = tid; s < 1024; s += 256) {       // 4 f32 per slot
                int r = s >> 3, c = (s & 7) << 2;
                f32x4 f = *(const f32x4*)(A + (size_t)(m0 + r) * K + k0 + c);
                As[buf][r][c + 0] = (__bf16)f[0];
                As[buf][r][c + 1] = (__bf16)f[1];
                As[buf][r][c + 2] = (__bf16)f[2];
                As[buf][r][c + 3] = (__bf16)f[3];
            }
        }
        for (int s = tid; s < 1024; s += 256) {
            int kk = s >> 5, nn = (s & 31) << 2;
            f32x4 f = *(const f32x4*)(Bp + (size_t)(k0 + kk) * ldb + n0 + nn);
            Bs[buf][nn + 0][kk] = (__bf16)f[0];
            Bs[buf][nn + 1][kk] = (__bf16)f[1];
            Bs[buf][nn + 2][kk] = (__bf16)f[2];
            Bs[buf][nn + 3][kk] = (__bf16)f[3];
        }
    };

    f32x8 acc[4][2];
    for (int i = 0; i < 4; ++i)
        for (int j = 0; j < 2; ++j) acc[i][j] = zero8();

    stage(0, 0);
    __syncthreads();
    int cur = 0;
    for (int k0 = 0; k0 < K; k0 += 32) {
        if (k0 + 32 < K) stage(k0 + 32, cur ^ 1);   // overlap with WMMA below

        bf16x16 af[4], bfr[2];
        for (int i = 0; i < 4; ++i)
            af[i] = ld_frag(&As[cur][wm * 64 + i * 16 + ln16][0], half);
        for (int j = 0; j < 2; ++j)
            bfr[j] = ld_frag(&Bs[cur][wn * 32 + j * 16 + ln16][0], half);
        for (int i = 0; i < 4; ++i)
            for (int j = 0; j < 2; ++j)
                acc[i][j] = wmma_bf16(af[i], bfr[j], acc[i][j]);

        __syncthreads();   // next buffer staged; current buffer free
        cur ^= 1;
    }

    // C tile layout: lane holds col N=ln16, rows M = r + 8*half.
    for (int i = 0; i < 4; ++i)
        for (int j = 0; j < 2; ++j)
            for (int r = 0; r < 8; ++r) {
                int row = m0 + wm * 64 + i * 16 + r + half * 8;
                int col = n0 + wn * 32 + j * 16 + ln16;
                Cp[(size_t)row * ldc + col] = acc[i][j][r];
            }
}

// ---------------------------------------------------------------------------
// RoPE + repack: qkv f32 [S][6144] ->
//   Qb bf16 [NH][S][HD], Kb bf16 [NKV][S][HD], Vt bf16 [NKV][HD][S] (transposed)
// ---------------------------------------------------------------------------
__global__ __launch_bounds__(256) void rope_pack_kernel(
    const float* __restrict__ qkv, __bf16* __restrict__ Qb,
    __bf16* __restrict__ Kb, __bf16* __restrict__ Vt) {
    const int COLS = NH * 64 + NKV * 64 + NKV * HD;   // 3584
    int idx = blockIdx.x * 256 + threadIdx.x;
    if (idx >= S_LEN * COLS) return;
    int pos = idx / COLS;
    int c   = idx % COLS;
    const float* row = qkv + (size_t)pos * NQKV;

    if (c < NH * 64) {                       // Q rotary pair (d, d+64)
        int hh = c >> 6, i = c & 63;
        float inv = __powf(10000.f, -(float)(2 * i) * (1.f / 128.f));
        float sn, cs;
        __sincosf((float)pos * inv, &sn, &cs);
        float x1 = row[hh * HD + i];
        float x2 = row[hh * HD + 64 + i];
        size_t o = ((size_t)hh * S_LEN + pos) * HD;
        Qb[o + i]      = (__bf16)(x1 * cs - x2 * sn);
        Qb[o + 64 + i] = (__bf16)(x2 * cs + x1 * sn);
    } else if (c < NH * 64 + NKV * 64) {     // K rotary pair
        int cc = c - NH * 64;
        int hh = cc >> 6, i = cc & 63;
        float inv = __powf(10000.f, -(float)(2 * i) * (1.f / 128.f));
        float sn, cs;
        __sincosf((float)pos * inv, &sn, &cs);
        float x1 = row[NH * HD + hh * HD + i];
        float x2 = row[NH * HD + hh * HD + 64 + i];
        size_t o = ((size_t)hh * S_LEN + pos) * HD;
        Kb[o + i]      = (__bf16)(x1 * cs - x2 * sn);
        Kb[o + 64 + i] = (__bf16)(x2 * cs + x1 * sn);
    } else {                                 // V: transpose into [kv][d][pos]
        int cc = c - (NH * 64 + NKV * 64);
        int hh = cc >> 7, d = cc & 127;
        float v = row[NH * HD + NKV * HD + hh * HD + d];
        Vt[((size_t)hh * HD + d) * S_LEN + pos] = (__bf16)v;
    }
}

// ---------------------------------------------------------------------------
// Causal flash attention. Grid (qtile=32, head=32), 128 threads = 4 waves.
// Each wave owns 16 query rows. K/V tiles (64 keys) double-buffered in LDS,
// prefetched with async global->LDS loads (16 async ops per wave per tile),
// waiting only on the older batch (asynccnt<=16) so copy overlaps compute.
// ---------------------------------------------------------------------------
__global__ __launch_bounds__(128) void flash_attn_kernel(
    const __bf16* __restrict__ Qb, const __bf16* __restrict__ Kb,
    const __bf16* __restrict__ Vt, __bf16* __restrict__ Ob) {
    __shared__ __bf16 Ks[2][64][128];   // [buf][key][d]   2 x 16 KB
    __shared__ __bf16 Vs[2][128][64];   // [buf][d][key]   2 x 16 KB
    __shared__ __bf16 Ps[4][16][64];    // per-wave P staging, 8 KB

    const int tid   = threadIdx.x;
    const int lane  = tid & 31;
    const int wid   = tid >> 5;
    const int half  = lane >> 4;
    const int ln16  = lane & 15;
    const int qtile = blockIdx.x;
    const int h     = blockIdx.y;
    const int hkv   = h >> 2;           // H/KV = 4
    const int q0    = qtile * 64 + wid * 16;

    const uint32_t ks_base = (uint32_t)(uintptr_t)&Ks[0][0][0];
    const uint32_t vs_base = (uint32_t)(uintptr_t)&Vs[0][0][0];

    auto issue_tile = [&](int t, int buf) {
        const uint32_t kb = ks_base + (uint32_t)buf * 16384u;
        for (int z = tid; z < 1024; z += 128) {          // 64x128 bf16
            uint32_t voff = (uint32_t)(
                (((size_t)hkv * S_LEN + t * 64 + (z >> 4)) * HD +
                 ((z & 15) << 3)) * 2);
            async_ld_b128(kb + z * 16, Kb, voff);
        }
        const uint32_t vb = vs_base + (uint32_t)buf * 16384u;
        for (int z = tid; z < 1024; z += 128) {          // 128x64 bf16
            uint32_t voff = (uint32_t)(
                (((size_t)hkv * HD + (z >> 3)) * S_LEN + t * 64 +
                 ((z & 7) << 3)) * 2);
            async_ld_b128(vb + z * 16, Vt, voff);
        }
    };

    // Q fragments for this wave's 16 rows: 4 K-steps over D=128.
    bf16x16 qf[4];
    {
        const __bf16* qp = Qb + ((size_t)h * S_LEN + q0 + ln16) * HD;
        for (int kt = 0; kt < 4; ++kt) qf[kt] = ld_frag(qp + kt * 32, half);
    }

    f32x8 oacc[8];
    for (int j = 0; j < 8; ++j) oacc[j] = zero8();
    float mrun[8], lrun[8];
    for (int r = 0; r < 8; ++r) { mrun[r] = -1e30f; lrun[r] = 0.f; }
    // (1/sqrt(128)) * log2(e)  -> softmax in exp2 domain
    const float CS = 0.08838834764831845f * 1.4426950408889634f;

    issue_tile(0, 0);
    int cur = 0;
    for (int t = 0; t <= qtile; ++t) {
        if (t < qtile) {
            issue_tile(t + 1, cur ^ 1);      // prefetch next tile
            asm volatile("s_wait_asynccnt 16" ::: "memory");
        } else {
            asm volatile("s_wait_asynccnt 0" ::: "memory");
        }
        __syncthreads();   // tile t visible to all waves

        // ---- S = Q * K^T : 4 n-tiles x (2x2 batched) k-steps ----
        f32x8 sacc[4];
        for (int nt = 0; nt < 4; ++nt) sacc[nt] = zero8();
        for (int nt = 0; nt < 4; ++nt) {
            const __bf16* krow = &Ks[cur][nt * 16 + ln16][0];
            for (int kp = 0; kp < 2; ++kp) {
                bf16x16 kb0 = ld_frag(krow + kp * 64, half);
                bf16x16 kb1 = ld_frag(krow + kp * 64 + 32, half);
                sacc[nt] = wmma_bf16(qf[2 * kp + 0], kb0, sacc[nt]);
                sacc[nt] = wmma_bf16(qf[2 * kp + 1], kb1, sacc[nt]);
            }
        }

        if (t == qtile) {   // causal mask on the diagonal tile
            for (int nt = 0; nt < 4; ++nt)
                for (int r = 0; r < 8; ++r) {
                    int q = q0 + r + half * 8;
                    int k = t * 64 + nt * 16 + ln16;
                    if (k > q) sacc[nt][r] = -1e30f;
                }
        }

        // ---- online softmax (row = r + 8*half, cols across 16 lanes) ----
        for (int r = 0; r < 8; ++r) {
            float mx = sacc[0][r];
            for (int nt = 1; nt < 4; ++nt) mx = fmaxf(mx, sacc[nt][r]);
            for (int d = 1; d < 16; d <<= 1)
                mx = fmaxf(mx, __shfl_xor(mx, d, 32));
            float mnew  = fmaxf(mrun[r], mx);
            float alpha = exp2f((mrun[r] - mnew) * CS);
            mrun[r] = mnew;
            float rs = 0.f;
            for (int nt = 0; nt < 4; ++nt) {
                float p = exp2f((sacc[nt][r] - mnew) * CS);
                sacc[nt][r] = p;
                rs += p;
            }
            for (int d = 1; d < 16; d <<= 1) rs += __shfl_xor(rs, d, 32);
            lrun[r] = lrun[r] * alpha + rs;
            for (int j = 0; j < 8; ++j) oacc[j][r] *= alpha;
        }

        // ---- reshape P (C layout -> A layout) through wave-private LDS ----
        for (int nt = 0; nt < 4; ++nt)
            for (int r = 0; r < 8; ++r)
                Ps[wid][r + half * 8][nt * 16 + ln16] = (__bf16)sacc[nt][r];
        asm volatile("" ::: "memory");   // keep ds writes before ds reads

        // ---- O += P * V : 8 d-tiles x 2 batched k-steps ----
        bf16x16 pf0 = ld_frag(&Ps[wid][ln16][0], half);
        bf16x16 pf1 = ld_frag(&Ps[wid][ln16][32], half);
        for (int j = 0; j < 8; ++j) {
            const __bf16* vrow = &Vs[cur][j * 16 + ln16][0];
            bf16x16 vb0 = ld_frag(vrow, half);
            bf16x16 vb1 = ld_frag(vrow + 32, half);
            oacc[j] = wmma_bf16(pf0, vb0, oacc[j]);
            oacc[j] = wmma_bf16(pf1, vb1, oacc[j]);
        }

        __syncthreads();   // all waves done with buffer `cur`
        cur ^= 1;
    }

    // ---- normalize and write O as bf16 [S][DMODEL] for the final GEMM ----
    float invl[8];
    for (int r = 0; r < 8; ++r) invl[r] = 1.0f / lrun[r];
    for (int j = 0; j < 8; ++j)
        for (int r = 0; r < 8; ++r) {
            int q = q0 + r + half * 8;
            int d = j * 16 + ln16;
            Ob[(size_t)q * DMODEL + h * HD + d] = (__bf16)(oacc[j][r] * invl[r]);
        }
}

// ---------------------------------------------------------------------------
// Host-side launcher
// ---------------------------------------------------------------------------
extern "C" void kernel_launch(void* const* d_in, const int* in_sizes, int n_in,
                              void* d_out, int out_size, void* d_ws,
                              size_t ws_size, hipStream_t stream) {
    (void)in_sizes; (void)n_in; (void)out_size; (void)ws_size;

    const float* x  = (const float*)d_in[0];   // [2048][4096]
    const float* Wq = (const float*)d_in[1];   // [4096][4096]
    const float* Wk = (const float*)d_in[2];   // [4096][1024]
    const float* Wv = (const float*)d_in[3];   // [4096][1024]
    const float* Wo = (const float*)d_in[4];   // [4096][4096]
    float* out = (float*)d_out;                // [2048][4096]

    char* ws = (char*)d_ws;
    float*  qkv = (float*)ws;  ws += (size_t)S_LEN * NQKV * sizeof(float);
    __bf16* Qb  = (__bf16*)ws; ws += (size_t)NH  * S_LEN * HD * sizeof(__bf16);
    __bf16* Kb  = (__bf16*)ws; ws += (size_t)NKV * S_LEN * HD * sizeof(__bf16);
    __bf16* Vt  = (__bf16*)ws; ws += (size_t)NKV * S_LEN * HD * sizeof(__bf16);
    __bf16* Ob  = (__bf16*)ws;

    // QKV projections into fused [S][6144] buffer (column offsets 0/4096/5120)
    gemm_kernel<false><<<dim3(32, 16), 256, 0, stream>>>(x, Wq, qkv,
                                                         DMODEL, 4096, NQKV);
    gemm_kernel<false><<<dim3(8, 16), 256, 0, stream>>>(x, Wk, qkv + 4096,
                                                        DMODEL, 1024, NQKV);
    gemm_kernel<false><<<dim3(8, 16), 256, 0, stream>>>(x, Wv, qkv + 5120,
                                                        DMODEL, 1024, NQKV);
    // RoPE + bf16 repack (V transposed for async-LDS-friendly attention)
    rope_pack_kernel<<<(S_LEN * 3584 + 255) / 256, 256, 0, stream>>>(qkv, Qb,
                                                                     Kb, Vt);
    // Causal flash attention
    flash_attn_kernel<<<dim3(32, 32), 128, 0, stream>>>(Qb, Kb, Vt, Ob);
    // Output projection
    gemm_kernel<true><<<dim3(32, 16), 256, 0, stream>>>(Ob, Wo, out,
                                                        DMODEL, 4096, DMODEL);
}
